// RobertaSelfAttention_39977555591806
// MI455X (gfx1250) — matrix operand order repaired and checked by reading, not compiled, not run on hardware
//
#include <hip/hip_runtime.h>

typedef __attribute__((ext_vector_type(16))) _Float16 v16h;
typedef __attribute__((ext_vector_type(8)))  _Float16 v8h;
typedef __attribute__((ext_vector_type(8)))  float    v8f;

#define HID   768
#define NHEAD 12
#define SEQ   256
#define HDIM  64
#define BATCH 32

static __device__ __forceinline__ v8f wmma_f16(v16h a, v16h b, v8f c) {
  // D(16x16,f32) = A(16x32,f16) * B(32x16,f16) + C
  return __builtin_amdgcn_wmma_f32_16x16x32_f16(false, a, false, b, (short)0, c, false, false);
}

// Build a 16-half WMMA operand from two contiguous 16B LDS runs.
static __device__ __forceinline__ v16h frag2(const _Float16* p0, const _Float16* p1) {
  v8h lo = *(const v8h*)p0;
  v8h hi = *(const v8h*)p1;
  return __builtin_shufflevector(lo, hi, 0, 1, 2, 3, 4, 5, 6, 7,
                                         8, 9, 10, 11, 12, 13, 14, 15);
}

// ---------------------------------------------------------------------------
// Kernel 1: fused QKV projection.  X[8192,768] * W[768,768] + b -> f16 out in
// per-head layout [B, NH, S, HD].  blockIdx.z selects Q/K/V.
// Block: 256 threads (8 waves), computes a 128x64 output tile; each wave owns
// a 32x32 subtile (2 A-frags x 2 B-frags = 4 WMMAs per k-step, 2 LDS
// b128 loads per WMMA).  Xs row-major, Wt transposed so both operands are
// contiguous 16B fragment runs.
// ---------------------------------------------------------------------------
__global__ __launch_bounds__(256)
void qkv_gemm_kernel(const float* __restrict__ X,
                     const float* __restrict__ Wq, const float* __restrict__ Wk,
                     const float* __restrict__ Wv,
                     const float* __restrict__ bq, const float* __restrict__ bk,
                     const float* __restrict__ bv,
                     _Float16* __restrict__ Qo, _Float16* __restrict__ Ko,
                     _Float16* __restrict__ Vo)
{
  __shared__ _Float16 Xs[128 * 40];  // [row][k], stride 80B: 16B-aligned, 20 banks
  __shared__ _Float16 Wt[64 * 40];   // [n][k] transposed, same stride

  const int tid = threadIdx.x;
  const int mt  = blockIdx.x;        // 0..63 (M tiles, M = B*S = 8192)
  const int nt  = blockIdx.y;        // 0..11 (N tiles, N = 768)
  const int mat = blockIdx.z;        // 0..2  (Q,K,V)
  const float* W    = (mat == 0) ? Wq : ((mat == 1) ? Wk : Wv);
  const float* bias = (mat == 0) ? bq : ((mat == 1) ? bk : bv);
  _Float16*    Out  = (mat == 0) ? Qo : ((mat == 1) ? Ko : Vo);

  const int m0 = mt * 128;
  const int n0 = nt * 64;

  const int wave   = tid >> 5;       // 0..7
  const int lane   = tid & 31;
  const int l16    = lane & 15;
  const int hi     = lane >> 4;      // lane half (per ISA fragment tables)
  const int rowGrp = wave >> 1;      // 0..3 : 32-row group
  const int colGrp = wave & 1;       // 0..1 : 32-col group

  // staging decomposition
  const int xr = tid >> 3;           // 0..31
  const int xc = (tid & 7) * 4;
  const int wr = tid >> 4;           // 0..15 (k)
  const int wc = (tid & 15) * 4;     // n

  v8f acc[2][2] = {};

  for (int k0 = 0; k0 < HID; k0 += 32) {
    __syncthreads();
    {
      // X tile 128x32, row-major in LDS (4 rows per thread)
#pragma unroll
      for (int i = 0; i < 4; ++i) {
        float4 x = *(const float4*)(X + (size_t)(m0 + xr + i * 32) * HID + k0 + xc);
        _Float16* p = &Xs[(xr + i * 32) * 40 + xc];
        p[0] = (_Float16)x.x; p[1] = (_Float16)x.y;
        p[2] = (_Float16)x.z; p[3] = (_Float16)x.w;
      }
      // W tile 32x64, transposed scatter into Wt[n][k]
      float4 w0 = *(const float4*)(W + (size_t)(k0 + wr)      * HID + n0 + wc);
      float4 w1 = *(const float4*)(W + (size_t)(k0 + wr + 16) * HID + n0 + wc);
      Wt[(wc + 0) * 40 + wr] = (_Float16)w0.x;
      Wt[(wc + 1) * 40 + wr] = (_Float16)w0.y;
      Wt[(wc + 2) * 40 + wr] = (_Float16)w0.z;
      Wt[(wc + 3) * 40 + wr] = (_Float16)w0.w;
      Wt[(wc + 0) * 40 + wr + 16] = (_Float16)w1.x;
      Wt[(wc + 1) * 40 + wr + 16] = (_Float16)w1.y;
      Wt[(wc + 2) * 40 + wr + 16] = (_Float16)w1.z;
      Wt[(wc + 3) * 40 + wr + 16] = (_Float16)w1.w;
    }
    if (k0 + 32 < HID)
      __builtin_prefetch(X + (size_t)(m0 + xr) * HID + k0 + 32 + xc, 0, 3);
    __syncthreads();

    // A fragments: halves 0-7 at K=hi*8, halves 8-15 at K=16+hi*8 (contiguous)
    const _Float16* ar0 = &Xs[(rowGrp * 32 +      l16) * 40];
    const _Float16* ar1 = &Xs[(rowGrp * 32 + 16 + l16) * 40];
    v16h a0 = frag2(ar0 + hi * 8, ar0 + 16 + hi * 8);
    v16h a1 = frag2(ar1 + hi * 8, ar1 + 16 + hi * 8);

    // B fragments from transposed Wt: halves 0-7 at K=hi*8, halves 8-15 at K=16+hi*8
    const _Float16* b0p = &Wt[(colGrp * 32 +      l16) * 40];
    const _Float16* b1p = &Wt[(colGrp * 32 + 16 + l16) * 40];
    v16h b0 = frag2(b0p + hi * 8, b0p + 16 + hi * 8);
    v16h b1 = frag2(b1p + hi * 8, b1p + 16 + hi * 8);

    acc[0][0] = wmma_f16(a0, b0, acc[0][0]);
    acc[0][1] = wmma_f16(a0, b1, acc[0][1]);
    acc[1][0] = wmma_f16(a1, b0, acc[1][0]);
    acc[1][1] = wmma_f16(a1, b1, acc[1][1]);
  }

  // epilogue: bias, convert to f16, scatter to per-head layout [B,NH,S,HD]
#pragma unroll
  for (int ri = 0; ri < 2; ++ri)
#pragma unroll
    for (int cj = 0; cj < 2; ++cj) {
      const int j    = n0 + colGrp * 32 + cj * 16 + l16;  // global output col
      const int head = j >> 6;
      const int d    = j & 63;
      const float bval = bias[j];
#pragma unroll
      for (int r = 0; r < 8; ++r) {
        const int gm = m0 + rowGrp * 32 + ri * 16 + r + hi * 8;  // row = b*S+s
        const int bb = gm >> 8;
        const int ss = gm & 255;
        Out[((size_t)(bb * NHEAD + head) * SEQ + ss) * HDIM + d] =
            (_Float16)(acc[ri][cj][r] + bval);
      }
    }
}

// ---------------------------------------------------------------------------
// Kernel 2: attention.  Block = 4 waves (128 thr) owns one (batch, head) and
// 64 query rows; each wave owns a 16-row q-tile with the full 256-wide score
// row resident in VGPRs (16 v8f accumulators).
// LDS: Vt (transposed V, persistent) + overlay { Qs,Ks | per-wave Ps }.
// ---------------------------------------------------------------------------
__global__ __launch_bounds__(128)
void attn_kernel(const _Float16* __restrict__ Q, const _Float16* __restrict__ K,
                 const _Float16* __restrict__ V,
                 const float* __restrict__ cmap, const float* __restrict__ amask,
                 const float* __restrict__ wcm_p, const float* __restrict__ bcm_p,
                 float* __restrict__ out)
{
  __shared__ __align__(16) unsigned char smem[51200];
  _Float16* Vt = (_Float16*)smem;               // [d=64][key 264]  33792B
  _Float16* Qs = (_Float16*)(smem + 33792);     // [64][72]  phase A (9216B)
  _Float16* Ks = (_Float16*)(smem + 43008);     // [16][72]  phase A (2304B)

  const int tid  = threadIdx.x;
  const int qt   = blockIdx.x;                  // 0..3
  const int head = blockIdx.y;                  // 0..11
  const int bb   = blockIdx.z;                  // 0..31
  const int wave = tid >> 5, lane = tid & 31;
  const int l16  = lane & 15, hi = lane >> 4;
  _Float16* Ps = (_Float16*)(smem + 33792 + wave * 4352);  // [16][136] phase B

  const size_t bh = (size_t)(bb * NHEAD + head) * SEQ * HDIM;
  const _Float16* Qg = Q + bh;
  const _Float16* Kg = K + bh;
  const _Float16* Vg = V + bh;
  const int q0 = qt * 64;
  const int qw = q0 + wave * 16;

  // stage the block's 64x64 Q tile (16B vectors, 144B row stride)
#pragma unroll
  for (int i = 0; i < 4; ++i) {
    const int idx = tid + i * 128;
    const int r = idx >> 3, c = (idx & 7) * 8;
    *(uint4*)(Qs + r * 72 + c) = *(const uint4*)(Qg + (q0 + r) * HDIM + c);
  }
  __syncthreads();

  // Q A-fragments (d = 0..31 and 32..63), built once, 2x b128 each
  const _Float16* qrowp = &Qs[(wave * 16 + l16) * 72];
  v16h aq0 = frag2(qrowp +      hi * 8, qrowp + 16 +      hi * 8);
  v16h aq1 = frag2(qrowp + 32 + hi * 8, qrowp + 32 + 16 + hi * 8);

  // ---- scores: full 16 x 256 row per wave -------------------------------
  v8f sc[16];
#pragma unroll
  for (int kt = 0; kt < 16; ++kt) {
    __syncthreads();
    {
      const int r = tid >> 3, c = (tid & 7) * 8;
      // K tile row-major (B-frag halves are d-contiguous)
      *(uint4*)(Ks + r * 72 + c) =
          *(const uint4*)(Kg + (kt * 16 + r) * HDIM + c);
      // V tile transposed into Vt[d][key] (so PV B-frags are key-contiguous)
      v8h hv = *(const v8h*)(Vg + (kt * 16 + r) * HDIM + c);
#pragma unroll
      for (int i = 0; i < 8; ++i)
        Vt[(c + i) * 264 + kt * 16 + r] = hv[i];
    }
    if (kt < 15)
      __builtin_prefetch(Kg + (kt + 1) * 16 * HDIM + lane * 8, 0, 3);
    __syncthreads();

    // B = K^T : lane = key col, halves = d (contiguous runs of 8)
    const _Float16* krow = &Ks[l16 * 72];
    v16h b0 = frag2(krow +      hi * 8, krow +      16 + hi * 8);
    v16h b1 = frag2(krow + 32 + hi * 8, krow + 32 + 16 + hi * 8);

    v8f c8 = {};
    c8 = wmma_f16(aq0, b0, c8);
    c8 = wmma_f16(aq1, b1, c8);
    sc[kt] = c8;
  }

  // ---- scale * contact-map + mask in exp2 domain, then row softmax -------
  const float LOG2E = 1.44269504088896340736f;
  const float wc2 = 0.125f * LOG2E * (*wcm_p);   // fold 1/sqrt(HD) and log2(e)
  const float bc2 = 0.125f * LOG2E * (*bcm_p);
  const float* cmb = cmap  + (size_t)bb * SEQ * SEQ;
  const float* amb = amask + (size_t)bb * SEQ;

#pragma unroll
  for (int kt = 0; kt < 16; ++kt) {
    const int kcol = kt * 16 + l16;
    const float amv2 = amb[kcol] * LOG2E;
#pragma unroll
    for (int r = 0; r < 8; ++r) {
      const int qrow = qw + r + hi * 8;
      const float cmv = fmaf(wc2, cmb[(size_t)qrow * SEQ + kcol], bc2);
      sc[kt][r] = fmaf(sc[kt][r], cmv, amv2);    // = log2e * (s*scale*cm + mask)
    }
  }

  float rmax[8], rinv[8];
#pragma unroll
  for (int r = 0; r < 8; ++r) {
    float m = sc[0][r];
#pragma unroll
    for (int kt = 1; kt < 16; ++kt) m = fmaxf(m, sc[kt][r]);
#pragma unroll
    for (int off = 1; off < 16; off <<= 1)
      m = fmaxf(m, __shfl_xor(m, off, 32));   // reduce within 16-lane half
    rmax[r] = m;
  }
  float rsum[8];
#pragma unroll
  for (int r = 0; r < 8; ++r) rsum[r] = 0.0f;
#pragma unroll
  for (int kt = 0; kt < 16; ++kt)
#pragma unroll
    for (int r = 0; r < 8; ++r) {
      const float e = exp2f(sc[kt][r] - rmax[r]);  // native v_exp_f32
      sc[kt][r] = e;
      rsum[r]  += e;
    }
#pragma unroll
  for (int r = 0; r < 8; ++r) {
    float s = rsum[r];
#pragma unroll
    for (int off = 1; off < 16; off <<= 1) s += __shfl_xor(s, off, 32);
    rinv[r] = 1.0f / s;
  }

  __syncthreads();   // all waves done with Qs/Ks before Ps overlays them

  // ---- ctx = P * V, two 128-key passes through per-wave Ps ---------------
  v8f ctx[4] = {};
#pragma unroll
  for (int p = 0; p < 2; ++p) {
    // write P: C-layout -> row-major [row][key] f16 in LDS (scalar transpose)
#pragma unroll
    for (int t = 0; t < 8; ++t) {
      const int kt = p * 8 + t;
#pragma unroll
      for (int r = 0; r < 8; ++r)
        Ps[(r + hi * 8) * 136 + t * 16 + l16] =
            (_Float16)(sc[kt][r] * rinv[r]);
    }
    // same-wave LDS RAW: DS ops are in-order per wave, no barrier needed
#pragma unroll
    for (int kk = 0; kk < 4; ++kk) {     // 4 x (K=32) chunks of this pass
      // A = P : lane = q-row, halves = key (contiguous runs of 8)
      const _Float16* prow = &Ps[l16 * 136 + kk * 32];
      v16h ap = frag2(prow + hi * 8, prow + 16 + hi * 8);
#pragma unroll
      for (int d2 = 0; d2 < 4; ++d2) {   // B = V : lane = d col, halves = key
        const _Float16* vrow =
            &Vt[(d2 * 16 + l16) * 264 + p * 128 + kk * 32];
        v16h bv = frag2(vrow + hi * 8, vrow + 16 + hi * 8);
        ctx[d2] = wmma_f16(ap, bv, ctx[d2]);
      }
    }
  }

  // ---- store ctx to [B, S, H] f32 ---------------------------------------
  float* outp = out + (size_t)bb * SEQ * HID + head * HDIM;
#pragma unroll
  for (int d2 = 0; d2 < 4; ++d2)
#pragma unroll
    for (int r = 0; r < 8; ++r) {
      const int qrow = qw + r + hi * 8;
      outp[(size_t)qrow * HID + d2 * 16 + l16] = ctx[d2][r];
    }
}

// ---------------------------------------------------------------------------
extern "C" void kernel_launch(void* const* d_in, const int* in_sizes, int n_in,
                              void* d_out, int out_size, void* d_ws, size_t ws_size,
                              hipStream_t stream)
{
  const float* X   = (const float*)d_in[0];
  const float* cm  = (const float*)d_in[1];
  const float* am  = (const float*)d_in[2];
  const float* Wq  = (const float*)d_in[3];
  const float* bq  = (const float*)d_in[4];
  const float* Wk  = (const float*)d_in[5];
  const float* bk  = (const float*)d_in[6];
  const float* Wv  = (const float*)d_in[7];
  const float* bv  = (const float*)d_in[8];
  const float* wcm = (const float*)d_in[9];
  const float* bcm = (const float*)d_in[10];
  float* out = (float*)d_out;

  const size_t perMat = (size_t)BATCH * NHEAD * SEQ * HDIM;  // 6,291,456 halves
  _Float16* Qf = (_Float16*)d_ws;
  _Float16* Kf = Qf + perMat;
  _Float16* Vf = Kf + perMat;   // total ~37.8 MB of workspace

  qkv_gemm_kernel<<<dim3(64, 12, 3), dim3(256), 0, stream>>>(
      X, Wq, Wk, Wv, bq, bk, bv, Qf, Kf, Vf);
  attn_kernel<<<dim3(4, 12, 32), dim3(128), 0, stream>>>(
      Qf, Kf, Vf, cm, am, wcm, bcm, out);
}
